// CoarseningRegularizerMx_50199577756210
// MI455X (gfx1250) — matrix operand order repaired and checked by reading, not compiled, not run on hardware
//
#include <hip/hip_runtime.h>

// Problem constants (match reference)
#define B_ROWS   8192
#define D_DIM    512
#define N_LABELS 256

// Tiling (fast path, double-buffered)
#define BM 64      // rows per workgroup (strip ownership -> deterministic row sums)
#define BN 128     // column tile (4 x 16 per wave -> amortize A fragments)
#define KC 32      // K chunk staged in LDS (one WMMA k-step)
#define LDK 40     // padded LDS row stride in bf16 (80B -> conflict-free b128 loads)
#define NCHUNK (D_DIM / KC)

// Fallback tiling (single-buffered, convert-in-kernel)
#define FB_KC  64
#define FB_LDK 72

typedef __attribute__((ext_vector_type(16))) __bf16 v16bf;
typedef __attribute__((ext_vector_type(8)))  float  v8f;
typedef __attribute__((ext_vector_type(4)))  int    v4i;
typedef unsigned long long uptr_t;

union Frag {
    v16bf v;
    uint4 u[2];
};

// Pointer flavors for the async-LDS builtin: global (AS1) / LDS (AS3) <4 x i32>*
typedef __attribute__((address_space(1))) v4i* gptr_v4i;
typedef __attribute__((address_space(3))) v4i* lptr_v4i;

#if defined(__has_builtin)
#if __has_builtin(__builtin_amdgcn_global_load_async_to_lds_b128)
#define HAVE_ASYNC_LDS 1
#endif
#if __has_builtin(__builtin_amdgcn_s_wait_asynccnt)
#define HAVE_WAIT_ASYNC 1
#endif
#endif

// 16B global -> LDS copy on the ASYNCcnt path (async data movement engine).
__device__ __forceinline__ void cp_async16(const void* g, void* l) {
#if defined(HAVE_ASYNC_LDS)
    __builtin_amdgcn_global_load_async_to_lds_b128(
        (gptr_v4i)(uptr_t)g, (lptr_v4i)(unsigned int)(uptr_t)l, 0, 0);
#else
    asm volatile("global_load_async_to_lds_b128 %0, %1, off"
                 :
                 : "v"((unsigned int)(uptr_t)l), "v"((gptr_v4i)(uptr_t)g)
                 : "memory");
#endif
}

#if defined(HAVE_WAIT_ASYNC)
#define WAIT_ASYNC(n) __builtin_amdgcn_s_wait_asynccnt(n)
#else
#define WAIT_ASYNC(n) asm volatile("s_wait_asynccnt %0" ::"i"(n) : "memory")
#endif

// round-to-nearest-even f32 -> bf16 (bit manipulation; no __bf16 arithmetic needed)
__device__ __forceinline__ unsigned short f2bf(float x) {
    unsigned int u = __float_as_uint(x);
    u += 0x7FFFu + ((u >> 16) & 1u);
    return (unsigned short)(u >> 16);
}
__device__ __forceinline__ float bf2f(unsigned short h) {
    return __uint_as_float(((unsigned int)h) << 16);
}
__device__ __forceinline__ unsigned int pack2(unsigned short a, unsigned short b) {
    return (unsigned int)a | ((unsigned int)b << 16);
}

// ---------------------------------------------------------------------------
// Prep kernel: sq[i] = ||z_i||^2 (wave per row), and label histogram.
// ---------------------------------------------------------------------------
__global__ __launch_bounds__(256) void prep_kernel(
    const float* __restrict__ z, const int* __restrict__ mx,
    float* __restrict__ sq, int* __restrict__ hist)
{
    const int lane = threadIdx.x & 31;
    const int wave = threadIdx.x >> 5;
    const int row  = blockIdx.x * 8 + wave;

    const float4* zr = (const float4*)(z + (size_t)row * D_DIM);
    float s = 0.f;
#pragma unroll
    for (int c = 0; c < D_DIM / 4 / 32; ++c) {
        float4 v = zr[lane + 32 * c];
        s += v.x * v.x + v.y * v.y + v.z * v.z + v.w * v.w;
    }
#pragma unroll
    for (int off = 16; off; off >>= 1) s += __shfl_xor(s, off, 32);
    if (lane == 0) sq[row] = s;

    const int tid = blockIdx.x * blockDim.x + threadIdx.x;
    if (tid < B_ROWS) atomicAdd(&hist[mx[tid]], 1);
}

// ---------------------------------------------------------------------------
// One-shot f32 -> split bf16 (hi, lo) conversion of the whole z matrix.
// ---------------------------------------------------------------------------
__global__ __launch_bounds__(256) void convert_kernel(
    const float* __restrict__ z,
    unsigned short* __restrict__ zhi, unsigned short* __restrict__ zlo)
{
    const size_t t = (size_t)blockIdx.x * 256 + threadIdx.x;  // two float4 each
    const float4* zf = (const float4*)z;
#pragma unroll
    for (int q = 0; q < 2; ++q) {
        const size_t idx = t * 2 + q;
        const float4 a = zf[idx];
        const unsigned short h0 = f2bf(a.x), h1 = f2bf(a.y), h2 = f2bf(a.z), h3 = f2bf(a.w);
        const unsigned short l0 = f2bf(a.x - bf2f(h0)), l1 = f2bf(a.y - bf2f(h1));
        const unsigned short l2 = f2bf(a.z - bf2f(h2)), l3 = f2bf(a.w - bf2f(h3));
        uint2 hh, ll;
        hh.x = pack2(h0, h1); hh.y = pack2(h2, h3);
        ll.x = pack2(l0, l1); ll.y = pack2(l2, l3);
        *(uint2*)(zhi + idx * 4) = hh;
        *(uint2*)(zlo + idx * 4) = ll;
    }
}

// Issue one K-chunk's async copies: 6 GLOBAL_LOAD_ASYNC_TO_LDS_B128 per wave.
__device__ __forceinline__ void issue_chunk(
    const unsigned short* __restrict__ zhi, const unsigned short* __restrict__ zlo,
    unsigned short* Ahi, unsigned short* Alo,
    unsigned short* Bhi, unsigned short* Blo,
    int tid, int i0, int j0, int k0)
{
    // A tile: 64 rows x 32 bf16 = 256 x 16B chunks, 1 per thread (x hi/lo).
    {
        const int r = tid >> 2;
        const int c = (tid & 3) << 3;                 // ushort units
        const size_t g = (size_t)(i0 + r) * D_DIM + k0 + c;
        const int    l = r * LDK + c;
        cp_async16(zhi + g, &Ahi[l]);
        cp_async16(zlo + g, &Alo[l]);
    }
    // B tile: 128 rows x 32 bf16 = 512 x 16B chunks, 2 per thread (x hi/lo).
#pragma unroll
    for (int it = 0; it < 2; ++it) {
        const int p = tid + it * 256;
        const int r = p >> 2;
        const int c = (p & 3) << 3;
        const size_t g = (size_t)(j0 + r) * D_DIM + k0 + c;
        const int    l = r * LDK + c;
        cp_async16(zhi + g, &Bhi[l]);
        cp_async16(zlo + g, &Blo[l]);
    }
}

// ---------------------------------------------------------------------------
// FAST path: split-bf16 WMMA GEMM, pre-converted operands, double-buffered
// async global->LDS staging (copy of chunk c+1 overlaps WMMAs of chunk c).
// dot(z_i,z_j) ~= hi.hi + hi.lo + lo.hi   (3x v_wmma_f32_16x16x32_bf16 per K=32)
// ---------------------------------------------------------------------------
__global__ __launch_bounds__(256) void pairdist_fast_kernel(
    const unsigned short* __restrict__ zhi, const unsigned short* __restrict__ zlo,
    const int* __restrict__ mx, const float* __restrict__ sq,
    const int* __restrict__ hist, float* __restrict__ out)
{
    __shared__ unsigned short Ahi[2][BM * LDK];
    __shared__ unsigned short Alo[2][BM * LDK];
    __shared__ unsigned short Bhi[2][BN * LDK];
    __shared__ unsigned short Blo[2][BN * LDK];
    __shared__ float lds_sum[BM];

    const int tid  = threadIdx.x;
    const int lane = tid & 31;
    const int wave = tid >> 5;
    const int wm   = wave & 3;   // M sub-tile (4 x 16 rows = 64)
    const int wn   = wave >> 2;  // N half   (2 x 64 cols = 128)
    const int i0   = blockIdx.x * BM;

    if (tid < BM) lds_sum[tid] = 0.f;

    const int mrow_base = wm * 16 + ((lane >> 4) << 3);
    float si[8];
    int   li[8];
#pragma unroll
    for (int v = 0; v < 8; ++v) {
        const int gi = i0 + mrow_base + v;
        si[v] = sq[gi];
        li[v] = mx[gi];
    }

    float rsum[8];
#pragma unroll
    for (int v = 0; v < 8; ++v) rsum[v] = 0.f;

    const v8f vzero = {};

    // Per-lane fragment offsets (fixed per wave)
    const int arow = wm * 16 + (lane & 15);
    const int koff = (lane >> 4) << 3;   // A: two 8-elem K runs at koff, koff+16
    const int kb   = (lane >> 4) << 3;   // B: two 8-elem K runs at kb, kb+16

    for (int j0 = 0; j0 < B_ROWS; j0 += BN) {
        v8f acc[4];
#pragma unroll
        for (int t = 0; t < 4; ++t) acc[t] = vzero;

        // Prologue: stage chunk 0 into buffer 0.
        issue_chunk(zhi, zlo, Ahi[0], Alo[0], Bhi[0], Blo[0], tid, i0, j0, 0);

        for (int c = 0; c < NCHUNK; ++c) {
            const int cur = c & 1;
            if (c + 1 < NCHUNK) {
                // Overlap: DMA chunk c+1 into the other buffer while computing c.
                issue_chunk(zhi, zlo, Ahi[1 - cur], Alo[1 - cur],
                            Bhi[1 - cur], Blo[1 - cur], tid, i0, j0, (c + 1) * KC);
                WAIT_ASYNC(6);   // only chunk c's 6 older async ops must be done
            } else {
                WAIT_ASYNC(0);
            }
            __syncthreads();     // chunk c visible to all waves

            // A fragment (16x32 bf16): two contiguous 8-elem K runs per lane
            Frag ahi, alo;
            ahi.u[0] = *(const uint4*)&Ahi[cur][arow * LDK + koff];
            ahi.u[1] = *(const uint4*)&Ahi[cur][arow * LDK + koff + 16];
            alo.u[0] = *(const uint4*)&Alo[cur][arow * LDK + koff];
            alo.u[1] = *(const uint4*)&Alo[cur][arow * LDK + koff + 16];

#pragma unroll
            for (int t = 0; t < 4; ++t) {
                // B fragment (32x16 bf16): two contiguous 8-elem K runs per lane
                const int brow = wn * 64 + t * 16 + (lane & 15);
                Frag bh, bl;
                bh.u[0] = *(const uint4*)&Bhi[cur][brow * LDK + kb];
                bh.u[1] = *(const uint4*)&Bhi[cur][brow * LDK + kb + 16];
                bl.u[0] = *(const uint4*)&Blo[cur][brow * LDK + kb];
                bl.u[1] = *(const uint4*)&Blo[cur][brow * LDK + kb + 16];

                acc[t] = __builtin_amdgcn_wmma_f32_16x16x32_bf16(false, ahi.v, false, bh.v,
                                                                 (short)0, acc[t], false, false);
                acc[t] = __builtin_amdgcn_wmma_f32_16x16x32_bf16(false, ahi.v, false, bl.v,
                                                                 (short)0, acc[t], false, false);
                acc[t] = __builtin_amdgcn_wmma_f32_16x16x32_bf16(false, alo.v, false, bh.v,
                                                                 (short)0, acc[t], false, false);
            }
            __syncthreads();     // all waves done reading before buffer reuse
        }

        // Epilogue: d2 = |zi|^2 + |zj|^2 - 2*dot ; mask = same label && i!=j
        const int ncol = lane & 15;
#pragma unroll
        for (int t = 0; t < 4; ++t) {
            const int   jg  = j0 + wn * 64 + t * 16 + ncol;
            const float sqj = sq[jg];
            const int   lj  = mx[jg];
#pragma unroll
            for (int v = 0; v < 8; ++v) {
                const int gi = i0 + mrow_base + v;
                const float d2 = si[v] + sqj - 2.f * acc[t][v];
                const float d  = d2 > 0.f ? __builtin_sqrtf(d2) : 0.f;
                rsum[v] += ((li[v] == lj) && (gi != jg)) ? d : 0.f;
            }
        }
    }

    // Reduce each row over its 16 lanes, then two commutative ds_add_f32 per row.
#pragma unroll
    for (int v = 0; v < 8; ++v) {
        float s = rsum[v];
        s += __shfl_xor(s, 1, 32);
        s += __shfl_xor(s, 2, 32);
        s += __shfl_xor(s, 4, 32);
        s += __shfl_xor(s, 8, 32);
        if ((lane & 15) == 0) atomicAdd(&lds_sum[mrow_base + v], s);
    }
    __syncthreads();

    if (tid < BM) {
        const int gi  = i0 + tid;
        const float cnt = (float)(hist[mx[gi]] - 1);
        out[gi] = cnt > 0.f ? lds_sum[tid] / cnt : 0.f;
    }
}

// ---------------------------------------------------------------------------
// FALLBACK path (small workspace): convert-in-kernel, BN=64, single buffer.
// ---------------------------------------------------------------------------
__global__ __launch_bounds__(256) void pairdist_fb_kernel(
    const float* __restrict__ z, const int* __restrict__ mx,
    const float* __restrict__ sq, const int* __restrict__ hist,
    float* __restrict__ out)
{
    __shared__ unsigned short Ahi[BM * FB_LDK];
    __shared__ unsigned short Alo[BM * FB_LDK];
    __shared__ unsigned short Bhi[64 * FB_LDK];
    __shared__ unsigned short Blo[64 * FB_LDK];
    __shared__ float lds_sum[BM];

    const int tid  = threadIdx.x;
    const int lane = tid & 31;
    const int wave = tid >> 5;
    const int wm   = wave & 3;
    const int wn   = wave >> 2;
    const int i0   = blockIdx.x * BM;

    if (tid < BM) lds_sum[tid] = 0.f;

    const int mrow_base = wm * 16 + ((lane >> 4) << 3);
    float si[8];
    int   li[8];
#pragma unroll
    for (int v = 0; v < 8; ++v) {
        const int gi = i0 + mrow_base + v;
        si[v] = sq[gi];
        li[v] = mx[gi];
    }
    float rsum[8];
#pragma unroll
    for (int v = 0; v < 8; ++v) rsum[v] = 0.f;

    const v8f vzero = {};

    for (int j0 = 0; j0 < B_ROWS; j0 += 64) {
        v8f acc0 = vzero, acc1 = vzero;

        for (int k0 = 0; k0 < D_DIM; k0 += FB_KC) {
            __syncthreads();
#pragma unroll
            for (int it = 0; it < (BM * FB_KC / 4) / 256; ++it) {
                const int p = tid + it * 256;
                const int r = p >> 4;
                const int c = (p & 15) << 2;

                const float4 a = *(const float4*)(z + (size_t)(i0 + r) * D_DIM + k0 + c);
                const float4 b = *(const float4*)(z + (size_t)(j0 + r) * D_DIM + k0 + c);
                if (k0 + FB_KC < D_DIM) {
                    __builtin_prefetch(z + (size_t)(j0 + r) * D_DIM + k0 + FB_KC + c, 0, 1);
                }
                unsigned short h0 = f2bf(a.x), h1 = f2bf(a.y), h2 = f2bf(a.z), h3 = f2bf(a.w);
                unsigned short l0 = f2bf(a.x - bf2f(h0)), l1 = f2bf(a.y - bf2f(h1));
                unsigned short l2 = f2bf(a.z - bf2f(h2)), l3 = f2bf(a.w - bf2f(h3));
                *(unsigned int*)&Ahi[r * FB_LDK + c]     = pack2(h0, h1);
                *(unsigned int*)&Ahi[r * FB_LDK + c + 2] = pack2(h2, h3);
                *(unsigned int*)&Alo[r * FB_LDK + c]     = pack2(l0, l1);
                *(unsigned int*)&Alo[r * FB_LDK + c + 2] = pack2(l2, l3);

                h0 = f2bf(b.x); h1 = f2bf(b.y); h2 = f2bf(b.z); h3 = f2bf(b.w);
                l0 = f2bf(b.x - bf2f(h0)); l1 = f2bf(b.y - bf2f(h1));
                l2 = f2bf(b.z - bf2f(h2)); l3 = f2bf(b.w - bf2f(h3));
                *(unsigned int*)&Bhi[r * FB_LDK + c]     = pack2(h0, h1);
                *(unsigned int*)&Bhi[r * FB_LDK + c + 2] = pack2(h2, h3);
                *(unsigned int*)&Blo[r * FB_LDK + c]     = pack2(l0, l1);
                *(unsigned int*)&Blo[r * FB_LDK + c + 2] = pack2(l2, l3);
            }
            __syncthreads();

#pragma unroll
            for (int ks = 0; ks < FB_KC; ks += 32) {
                const int arow = wm * 16 + (lane & 15);
                const int koff = ks + ((lane >> 4) << 3);
                Frag ahi, alo;
                ahi.u[0] = *(const uint4*)&Ahi[arow * FB_LDK + koff];
                ahi.u[1] = *(const uint4*)&Ahi[arow * FB_LDK + koff + 16];
                alo.u[0] = *(const uint4*)&Alo[arow * FB_LDK + koff];
                alo.u[1] = *(const uint4*)&Alo[arow * FB_LDK + koff + 16];

                const int kb = ks + ((lane >> 4) << 3);
                const int brow0 = wn * 32 + (lane & 15);
                const int brow1 = brow0 + 16;
                Frag b0h, b0l, b1h, b1l;
                b0h.u[0] = *(const uint4*)&Bhi[brow0 * FB_LDK + kb];
                b0h.u[1] = *(const uint4*)&Bhi[brow0 * FB_LDK + kb + 16];
                b0l.u[0] = *(const uint4*)&Blo[brow0 * FB_LDK + kb];
                b0l.u[1] = *(const uint4*)&Blo[brow0 * FB_LDK + kb + 16];
                b1h.u[0] = *(const uint4*)&Bhi[brow1 * FB_LDK + kb];
                b1h.u[1] = *(const uint4*)&Bhi[brow1 * FB_LDK + kb + 16];
                b1l.u[0] = *(const uint4*)&Blo[brow1 * FB_LDK + kb];
                b1l.u[1] = *(const uint4*)&Blo[brow1 * FB_LDK + kb + 16];

                acc0 = __builtin_amdgcn_wmma_f32_16x16x32_bf16(false, ahi.v, false, b0h.v,
                                                               (short)0, acc0, false, false);
                acc0 = __builtin_amdgcn_wmma_f32_16x16x32_bf16(false, ahi.v, false, b0l.v,
                                                               (short)0, acc0, false, false);
                acc0 = __builtin_amdgcn_wmma_f32_16x16x32_bf16(false, alo.v, false, b0h.v,
                                                               (short)0, acc0, false, false);
                acc1 = __builtin_amdgcn_wmma_f32_16x16x32_bf16(false, ahi.v, false, b1h.v,
                                                               (short)0, acc1, false, false);
                acc1 = __builtin_amdgcn_wmma_f32_16x16x32_bf16(false, ahi.v, false, b1l.v,
                                                               (short)0, acc1, false, false);
                acc1 = __builtin_amdgcn_wmma_f32_16x16x32_bf16(false, alo.v, false, b1h.v,
                                                               (short)0, acc1, false, false);
            }
        }

        const int ncol = lane & 15;
        const int j_a  = j0 + wn * 32 + ncol;
        const int j_b  = j_a + 16;
        const float sqa = sq[j_a], sqb = sq[j_b];
        const int   la  = mx[j_a], lb = mx[j_b];
#pragma unroll
        for (int v = 0; v < 8; ++v) {
            const int gi = i0 + mrow_base + v;
            float d2a = si[v] + sqa - 2.f * acc0[v];
            float da  = d2a > 0.f ? __builtin_sqrtf(d2a) : 0.f;
            rsum[v]  += ((li[v] == la) && (gi != j_a)) ? da : 0.f;
            float d2b = si[v] + sqb - 2.f * acc1[v];
            float db  = d2b > 0.f ? __builtin_sqrtf(d2b) : 0.f;
            rsum[v]  += ((li[v] == lb) && (gi != j_b)) ? db : 0.f;
        }
    }

#pragma unroll
    for (int v = 0; v < 8; ++v) {
        float s = rsum[v];
        s += __shfl_xor(s, 1, 32);
        s += __shfl_xor(s, 2, 32);
        s += __shfl_xor(s, 4, 32);
        s += __shfl_xor(s, 8, 32);
        if ((lane & 15) == 0) atomicAdd(&lds_sum[mrow_base + v], s);
    }
    __syncthreads();

    if (tid < BM) {
        const int gi  = i0 + tid;
        const float cnt = (float)(hist[mx[gi]] - 1);
        out[gi] = cnt > 0.f ? lds_sum[tid] / cnt : 0.f;
    }
}

// ---------------------------------------------------------------------------
extern "C" void kernel_launch(void* const* d_in, const int* in_sizes, int n_in,
                              void* d_out, int out_size, void* d_ws, size_t ws_size,
                              hipStream_t stream) {
    const float* z  = (const float*)d_in[0];
    const int*   mx = (const int*)d_in[1];
    float*       out = (float*)d_out;

    // ws layout: [sq: 8192 f32][hist: 256 i32][pad to 64KB][zhi: 8MB][zlo: 8MB]
    float* sq   = (float*)d_ws;
    int*   hist = (int*)((char*)d_ws + B_ROWS * sizeof(float));
    unsigned short* zhi = (unsigned short*)((char*)d_ws + (64 << 10));
    unsigned short* zlo = zhi + (size_t)B_ROWS * D_DIM;

    const size_t need = (64 << 10) + 2ull * B_ROWS * D_DIM * sizeof(unsigned short);

    (void)hipMemsetAsync(hist, 0, N_LABELS * sizeof(int), stream);
    prep_kernel<<<B_ROWS / 8, 256, 0, stream>>>(z, mx, sq, hist);

    if (ws_size >= need) {
        convert_kernel<<<(B_ROWS * D_DIM / 8) / 256, 256, 0, stream>>>(z, zhi, zlo);
        pairdist_fast_kernel<<<B_ROWS / BM, 256, 0, stream>>>(zhi, zlo, mx, sq, hist, out);
    } else {
        pairdist_fb_kernel<<<B_ROWS / BM, 256, 0, stream>>>(z, mx, sq, hist, out);
    }
}
